// MultiHeadAttention_9053791060340
// MI455X (gfx1250) — matrix operand order repaired and checked by reading, not compiled, not run on hardware
//
#include <hip/hip_runtime.h>
#include <hip/hip_bf16.h>

// ---------------- problem constants ----------------
constexpr int Bb = 4, Ss = 2048, DMm = 512, Hh = 8, HDd = 64;
constexpr float SCALE = 0.125f; // 1/sqrt(64)

// ---------------- workspace layout (bytes) ----------------
constexpr size_t XB_OFF = 0;          // bf16 x, 8 MiB (reused as ctx after qkv gemm)
constexpr size_t WB_OFF = 8388608;    // bf16 Wq|Wk|Wv|Wo, 2 MiB
constexpr size_t Q_OFF  = 10485760;   // Q  [B,H,S,HD] bf16, 8 MiB
constexpr size_t K_OFF  = 18874368;   // K  [B,H,S,HD] bf16, 8 MiB
constexpr size_t VT_OFF = 27262976;   // Vt [B,H,HD,S] bf16, 8 MiB

// ---------------- vector types ----------------
typedef __attribute__((ext_vector_type(8)))  float        v8f;
typedef __attribute__((ext_vector_type(16))) __bf16       v16bf;
typedef __attribute__((ext_vector_type(4)))  unsigned int v4u;
typedef __attribute__((ext_vector_type(4)))  unsigned int u32x4;
typedef __attribute__((ext_vector_type(8)))  int          i32x8;
typedef __attribute__((ext_vector_type(4)))  int          i32x4;

union BFPack {
    v16bf bf;
    struct { v4u lo; v4u hi; } s;
};

__device__ __forceinline__ unsigned short f2bf(float f) {
    unsigned int u = __float_as_uint(f);
    u += 0x7FFFu + ((u >> 16) & 1u);   // round-to-nearest-even
    return (unsigned short)(u >> 16);
}

// A-operand 16x32 bf16: lane<16 -> row, K chunks {0..7, 16..23}; lane>=16 -> K chunks {8..15, 24..31}
__device__ __forceinline__ v16bf load_a_tile(const unsigned short* base, int stride,
                                             int row, int k0, int hi16) {
    const unsigned short* p = base + row * stride + k0 + hi16 * 8;
    BFPack r;
    r.s.lo = *(const v4u*)p;
    r.s.hi = *(const v4u*)(p + 16);
    return r.bf;
}

// B-operand 32x16 bf16: lane<16 -> col, K rows {0..7, 16..23}; lane>=16 -> K rows {8..15, 24..31}
__device__ __forceinline__ v16bf load_b_tile(const unsigned short* base, int stride,
                                             int row, int k0, int hi16) {
    const unsigned short* p = base + row * stride + k0 + hi16 * 8;
    BFPack r;
    r.s.lo = *(const v4u*)p;
    r.s.hi = *(const v4u*)(p + 16);
    return r.bf;
}

__device__ __forceinline__ v8f wmma_bf16(v16bf a, v16bf b, v8f c) {
    return __builtin_amdgcn_wmma_f32_16x16x32_bf16(false, a, false, b, (short)0, c, false, false);
}

// ---- Tensor Data Mover: 2D tile load (bf16 elements) per CDNA5 ISA ----
// gptr points at tile start; dim0/dim1 = remaining tensor extent; stride0 in elements.
__device__ __forceinline__ void tdm_load_2d(const unsigned short* gptr, unsigned lds_off,
                                            unsigned dim0, unsigned dim1, unsigned stride0,
                                            unsigned tile0, unsigned tile1) {
    unsigned long long ga = (unsigned long long)(uintptr_t)gptr;
    u32x4 g0;
    g0[0] = 1u;                                                  // count=1, user mode, no gather
    g0[1] = lds_off;                                             // LDS byte address
    g0[2] = (unsigned)(ga & 0xFFFFFFFFu);                        // global_addr[31:0]
    g0[3] = (unsigned)((ga >> 32) & 0x01FFFFFFu) | (2u << 30);   // global_addr[56:32] | type=2
    i32x8 g1;
    g1[0] = 0x00010000;                                          // wg_mask=0, data_size=1 (2B)
    g1[1] = (int)((dim0 & 0xFFFFu) << 16);                       // tensor_dim0[15:0]
    g1[2] = (int)((dim0 >> 16) | ((dim1 & 0xFFFFu) << 16));      // dim0[31:16] | dim1[15:0]
    g1[3] = (int)((dim1 >> 16) | (tile0 << 16));                 // dim1[31:16] | tile_dim0
    g1[4] = (int)(tile1 & 0xFFFFu);                              // tile_dim1 | tile_dim2=0
    g1[5] = (int)stride0;                                        // tensor_dim0_stride[31:0]
    g1[6] = 0;                                                   // stride0[47:32] | stride1 lo (2D unused)
    g1[7] = 0;
    i32x4 gz = {0, 0, 0, 0};
#if __clang_major__ >= 23
    i32x8 gz8 = {0, 0, 0, 0, 0, 0, 0, 0};
    __builtin_amdgcn_tensor_load_to_lds(g0, g1, gz, gz, gz8, 0);
#else
    __builtin_amdgcn_tensor_load_to_lds(g0, g1, gz, gz, 0);
#endif
}

// ---------------- kernel 1: fp32 -> bf16 conversion ----------------
__global__ void convert_kernel(const float* __restrict__ x,
                               const float* __restrict__ Wq, const float* __restrict__ Wk,
                               const float* __restrict__ Wv, const float* __restrict__ Wo,
                               unsigned short* __restrict__ xb, unsigned short* __restrict__ wb) {
    const int NX = Bb * Ss * DMm;
    const int NW = DMm * DMm;
    int idx = blockIdx.x * blockDim.x + threadIdx.x;
    if (idx < NX) {
        xb[idx] = f2bf(x[idx]);
    } else {
        int j = idx - NX;
        if (j < 4 * NW) {
            int w = j / NW, o = j % NW;
            const float* src = (w == 0) ? Wq : (w == 1) ? Wk : (w == 2) ? Wv : Wo;
            wb[j] = f2bf(src[o]);
        }
    }
}

// ---------------- kernel 2: fused QKV projection GEMM ----------------
__global__ void qkv_gemm_kernel(const unsigned short* __restrict__ xb,
                                const unsigned short* __restrict__ wb,
                                const float* __restrict__ bq, const float* __restrict__ bk,
                                const float* __restrict__ bv,
                                unsigned short* __restrict__ Q, unsigned short* __restrict__ K,
                                unsigned short* __restrict__ Vt) {
    int lane = threadIdx.x & 31;
    int lr = lane & 15, hi = lane >> 4;
    int wid = blockIdx.x * 8 + (threadIdx.x >> 5);
    int mtile = wid / 24;
    int rem = wid % 24;
    int mat = rem >> 3;          // 0=Q 1=K 2=V
    int ntile = rem & 7;
    int m0 = mtile * 16, n0 = ntile * 64;
    const unsigned short* W = wb + mat * DMm * DMm;

    v8f acc[4] = {};
#pragma unroll 4
    for (int k0 = 0; k0 < DMm; k0 += 32) {
        v16bf a = load_a_tile(xb, DMm, m0 + lr, k0, hi);
#pragma unroll
        for (int t = 0; t < 4; ++t) {
            v16bf bm = load_b_tile(W, DMm, n0 + t * 16 + lr, k0, hi);
            acc[t] = wmma_bf16(a, bm, acc[t]);
        }
    }

    const float* bias = (mat == 0) ? bq : (mat == 1) ? bk : bv;
#pragma unroll
    for (int t = 0; t < 4; ++t) {
#pragma unroll
        for (int r = 0; r < 8; ++r) {
            int col = n0 + t * 16 + lr;
            int row = m0 + r + hi * 8;
            float v = acc[t][r] + bias[col];
            int b = row / Ss, s = row % Ss;
            int h = col >> 6, hd = col & 63;
            unsigned short bv16 = f2bf(v);
            if (mat == 2)
                Vt[((size_t)(b * Hh + h) * HDd + hd) * Ss + s] = bv16;
            else {
                unsigned short* dst = (mat == 0) ? Q : K;
                dst[((size_t)(b * Hh + h) * Ss + s) * HDd + hd] = bv16;
            }
        }
    }
}

// ---------------- kernel 3: flash attention, TDM-staged K/V in LDS ----------------
// 8 waves/block share (b,h); waves own consecutive 16-row q tiles (128 rows/block).
// Wave 0 drives double-buffered TDM loads of 32-key K (32x64) and V (64x32) tiles.
// grid = B*H*(S/128) = 512 blocks.
constexpr int KB0 = 0;       // ushort offsets inside smem
constexpr int KB1 = 2048;    // K tiles: 32x64
constexpr int VB0 = 4096;    // V tiles: 64x32
constexpr int VB1 = 6144;
constexpr int PB  = 8192;    // per-wave 16x48 P staging
__global__ void attention_kernel(const unsigned short* __restrict__ Q,
                                 const unsigned short* __restrict__ K,
                                 const unsigned short* __restrict__ Vt,
                                 const float* __restrict__ days,
                                 const float* __restrict__ rate_p,
                                 unsigned short* __restrict__ ctx) {
    __shared__ unsigned short smem[PB + 8 * 16 * 48];   // 28 KiB

    int lane = threadIdx.x & 31;
    int lr = lane & 15, hi = lane >> 4;
    int w = threadIdx.x >> 5;

    int bh = blockIdx.x >> 4;
    int qb = blockIdx.x & 15;
    int b = bh >> 3, h = bh & 7;
    int q0 = qb * 128 + w * 16;

    const float rate = rate_p[0];
    const unsigned short* Qbh = Q + (size_t)(b * Hh + h) * Ss * HDd;
    const unsigned short* Kbh = K + (size_t)(b * Hh + h) * Ss * HDd;
    const unsigned short* Vbh = Vt + (size_t)(b * Hh + h) * HDd * Ss;
    const float* days_b = days + b * Ss;
    unsigned smem_base = (unsigned)(uintptr_t)(void*)&smem[0];

    v16bf qa0 = load_a_tile(Qbh, HDd, q0 + lr, 0, hi);
    v16bf qa1 = load_a_tile(Qbh, HDd, q0 + lr, 32, hi);
    float tq[8];
#pragma unroll
    for (int r = 0; r < 8; ++r) tq[r] = days_b[q0 + r + hi * 8];

    float m[8], l[8];
#pragma unroll
    for (int r = 0; r < 8; ++r) { m[r] = -3.0e38f; l[r] = 0.0f; }
    v8f o[4] = {};

    int nkb = (qb + 1) * 4;    // block-uniform key-block count (keys 0 .. qb*128+127)

    // prologue: stage tile 0 into buffer 0
    if (w == 0) {
        tdm_load_2d(Kbh, smem_base + KB0 * 2, HDd, Ss, HDd, HDd, 32);
        tdm_load_2d(Vbh, smem_base + VB0 * 2, Ss, HDd, Ss, 32, HDd);
    }

    for (int kb = 0; kb < nkb; ++kb) {
        int j0 = kb * 32;
        int cur = kb & 1;
        if (w == 0) {
            if (kb + 1 < nkb) {
                int j1 = j0 + 32;
                unsigned koff = (kb & 1) ? KB0 : KB1;   // next buffer
                unsigned voff = (kb & 1) ? VB0 : VB1;
                tdm_load_2d(Kbh + (size_t)j1 * HDd, smem_base + koff * 2,
                            HDd, Ss - j1, HDd, HDd, 32);
                tdm_load_2d(Vbh + j1, smem_base + voff * 2,
                            Ss - j1, HDd, Ss, 32, HDd);
                __builtin_amdgcn_s_wait_tensorcnt(2);   // current tile's 2 TDM ops done
            } else {
                __builtin_amdgcn_s_wait_tensorcnt(0);
            }
        }
        __syncthreads();                                // current buffer visible to all waves

        if (j0 <= q0 + 15) {                            // causal: this wave needs this key block
            const unsigned short* ldsK = smem + (cur ? KB1 : KB0);   // 32x64, stride 64
            const unsigned short* ldsV = smem + (cur ? VB1 : VB0);   // 64x32, stride 32

            // ---- scores: S(16x32) = Q(16x64) . K^T, 4 WMMAs ----
            float st[2][8];
#pragma unroll
            for (int t = 0; t < 2; ++t) {
                v8f sa = {};
                v16bf kb0 = load_b_tile(ldsK, HDd, t * 16 + lr, 0, hi);
                v16bf kb1 = load_b_tile(ldsK, HDd, t * 16 + lr, 32, hi);
                sa = wmma_bf16(qa0, kb0, sa);
                sa = wmma_bf16(qa1, kb1, sa);

                int col = j0 + t * 16 + lr;
                int ck = (col < Ss) ? col : (Ss - 1);
                float tk = days_b[ck];
#pragma unroll
                for (int r = 0; r < 8; ++r) {
                    int row = q0 + r + hi * 8;
                    float sv = sa[r] * SCALE * __expf(-rate * fabsf(tq[r] - tk));
                    st[t][r] = (col > row) ? -3.0e38f : sv;
                }
            }

            // ---- online softmax: row stats across 16-lane halves ----
            unsigned short* pw = smem + PB + w * (16 * 48);
#pragma unroll
            for (int r = 0; r < 8; ++r) {
                float c = fmaxf(st[0][r], st[1][r]);
                c = fmaxf(c, __shfl_xor(c, 1));
                c = fmaxf(c, __shfl_xor(c, 2));
                c = fmaxf(c, __shfl_xor(c, 4));
                c = fmaxf(c, __shfl_xor(c, 8));
                float mnew = fmaxf(m[r], c);
                float alpha = __expf(m[r] - mnew);
                float p0 = __expf(st[0][r] - mnew);
                float p1 = __expf(st[1][r] - mnew);
                float sum = p0 + p1;
                sum += __shfl_xor(sum, 1);
                sum += __shfl_xor(sum, 2);
                sum += __shfl_xor(sum, 4);
                sum += __shfl_xor(sum, 8);
                l[r] = l[r] * alpha + sum;
                m[r] = mnew;
#pragma unroll
                for (int t2 = 0; t2 < 4; ++t2) o[t2][r] *= alpha;
                pw[(r + hi * 8) * 48 + lr]      = f2bf(p0);
                pw[(r + hi * 8) * 48 + 16 + lr] = f2bf(p1);
            }

            // ---- P(16x32) . V(32x64): 4 WMMAs from LDS ----
            v16bf pa = load_a_tile(pw, 48, lr, 0, hi);
#pragma unroll
            for (int t2 = 0; t2 < 4; ++t2) {
                v16bf vb = load_b_tile(ldsV, 32, t2 * 16 + lr, 0, hi);
                o[t2] = wmma_bf16(pa, vb, o[t2]);
            }
        }
        __syncthreads();                                // all readers done before buffer reuse
    }

    // ---- normalize and write ctx [B*S, DM] bf16 ----
#pragma unroll
    for (int r = 0; r < 8; ++r) {
        float inv = 1.0f / l[r];
        int row = b * Ss + q0 + r + hi * 8;
#pragma unroll
        for (int t2 = 0; t2 < 4; ++t2) {
            int col = h * HDd + t2 * 16 + lr;
            ctx[(size_t)row * DMm + col] = f2bf(o[t2][r] * inv);
        }
    }
}

// ---------------- kernel 4: output projection GEMM ----------------
__global__ void out_gemm_kernel(const unsigned short* __restrict__ ctx,
                                const unsigned short* __restrict__ wb,
                                const float* __restrict__ bo,
                                float* __restrict__ out) {
    int lane = threadIdx.x & 31;
    int lr = lane & 15, hi = lane >> 4;
    int wid = blockIdx.x * 8 + (threadIdx.x >> 5);
    int mtile = wid >> 3, ntile = wid & 7;
    int m0 = mtile * 16, n0 = ntile * 64;
    const unsigned short* Wo = wb + 3 * DMm * DMm;

    v8f acc[4] = {};
#pragma unroll 4
    for (int k0 = 0; k0 < DMm; k0 += 32) {
        v16bf a = load_a_tile(ctx, DMm, m0 + lr, k0, hi);
#pragma unroll
        for (int t = 0; t < 4; ++t) {
            v16bf bm = load_b_tile(Wo, DMm, n0 + t * 16 + lr, k0, hi);
            acc[t] = wmma_bf16(a, bm, acc[t]);
        }
    }

#pragma unroll
    for (int t = 0; t < 4; ++t) {
#pragma unroll
        for (int r = 0; r < 8; ++r) {
            int col = n0 + t * 16 + lr;
            int row = m0 + r + hi * 8;
            out[(size_t)row * DMm + col] = acc[t][r] + bo[col];
        }
    }
}

// ---------------- launcher ----------------
extern "C" void kernel_launch(void* const* d_in, const int* in_sizes, int n_in,
                              void* d_out, int out_size, void* d_ws, size_t ws_size,
                              hipStream_t stream) {
    const float* x    = (const float*)d_in[0];
    // d_in[1] = mask: causal tril, applied analytically (j <= i), never read
    const float* days = (const float*)d_in[2];
    const float* Wq   = (const float*)d_in[3];
    const float* bq   = (const float*)d_in[4];
    const float* Wk   = (const float*)d_in[5];
    const float* bk   = (const float*)d_in[6];
    const float* Wv   = (const float*)d_in[7];
    const float* bv   = (const float*)d_in[8];
    const float* Wo   = (const float*)d_in[9];
    const float* bo   = (const float*)d_in[10];
    const float* rate = (const float*)d_in[11];
    float* out = (float*)d_out;

    char* ws = (char*)d_ws;
    unsigned short* xb  = (unsigned short*)(ws + XB_OFF);
    unsigned short* wb  = (unsigned short*)(ws + WB_OFF);
    unsigned short* Qb  = (unsigned short*)(ws + Q_OFF);
    unsigned short* Kb  = (unsigned short*)(ws + K_OFF);
    unsigned short* Vtb = (unsigned short*)(ws + VT_OFF);
    unsigned short* ctxb = xb;

    convert_kernel<<<20480, 256, 0, stream>>>(x, Wq, Wk, Wv, Wo, xb, wb);
    qkv_gemm_kernel<<<1536, 256, 0, stream>>>(xb, wb, bq, bk, bv, Qb, Kb, Vtb);
    attention_kernel<<<512, 256, 0, stream>>>(Qb, Kb, Vtb, days, rate, ctxb);
    out_gemm_kernel<<<512, 256, 0, stream>>>(ctxb, wb, bo, out);
}
